// TransformerDSOEncoder_60859686584833
// MI455X (gfx1250) — matrix operand order repaired and checked by reading, not compiled, not run on hardware
//
#include <hip/hip_runtime.h>

#define B_   32
#define S_   256
#define D_   512
#define H_   8
#define L_   6
#define DFF_ 2048
#define V_   64
#define DH_  64
#define NTOK (B_ * S_)

typedef __attribute__((ext_vector_type(16))) __bf16 v16bf;
typedef __attribute__((ext_vector_type(8)))  float  v8f;

union Frag16 { v16bf v; unsigned short u[16]; };
union U4x8   { uint4 q; unsigned short s[8]; };

__device__ __forceinline__ unsigned short f2bf(float f) {
    unsigned int u = __float_as_uint(f);
    unsigned int r = u + 0x7FFFu + ((u >> 16) & 1u);   // round-to-nearest-even
    return (unsigned short)(r >> 16);
}

// ---------------------------------------------------------------------------
// Generic batched bf16 WMMA GEMM: C = act(alpha * A@B + bias + Cadd)
//   A: [M,K] bf16 row-major (per-batch offset via 2-level stride)
//   B: [K,N] bf16 row-major, or [N,K] row-major when TRANSB (i.e. B = srcT)
//   C: fp32 or bf16 output
// block = 256 threads (8 waves); tile 128x128, K-chunk 32.
// Wave w: rows [(w>>2)*64, +64), cols [(w&3)*32, +32) -> 4x2 WMMA tiles.
// Out-of-range rows/cols are address-clamped on load (values unused) and
// guarded only at the store -- no divergent branches in the hot loop.
// ---------------------------------------------------------------------------
#define TM 128
#define TN 128
#define TK 32
#define LST 40   // LDS row stride (ushorts); 80B = multiple of 16B for b128 ops

template<bool TRANSB, bool OUTBF16, bool RELU>
__global__ __launch_bounds__(256) void gemm_kernel(
    const unsigned short* __restrict__ A, const unsigned short* __restrict__ Bm,
    const float* __restrict__ bias, const float* __restrict__ Cadd,
    float* __restrict__ Cf, unsigned short* __restrict__ Cb,
    int M, int N, int K, int lda, int ldb, int ldc, int ldadd,
    int divA, long sA0, long sA1,
    int divB, long sB0, long sB1,
    int divC, long sC0, long sC1,
    int divAdd, long sAdd0, long sAdd1,
    float alpha)
{
    __shared__ __align__(16) unsigned short At[TM * LST];
    __shared__ __align__(16) unsigned short Bt[TN * LST];

    const int tid = threadIdx.x;
    const int w = tid >> 5;          // wave id 0..7
    const int l = tid & 31;          // lane id
    const int mBase = (w >> 2) * 64; // wave row block
    const int nBase = (w & 3) * 32;  // wave col block
    const int z = blockIdx.z;

    const long offA = (long)(z / divA) * sA0 + (long)(z % divA) * sA1;
    const long offB = (long)(z / divB) * sB0 + (long)(z % divB) * sB1;
    const long offC = (long)(z / divC) * sC0 + (long)(z % divC) * sC1;
    const long offAdd = (long)(z / divAdd) * sAdd0 + (long)(z % divAdd) * sAdd1;

    const int m0 = blockIdx.x * TM;
    const int n0 = blockIdx.y * TN;

    v8f acc[4][2] = { { {}, {} }, { {}, {} }, { {}, {} }, { {}, {} } };

    for (int k0 = 0; k0 < K; k0 += TK) {
        __syncthreads();
        // ---- stage A tile (128x32) as b128, address-clamped ----
        #pragma unroll
        for (int i = 0; i < 2; ++i) {
            int p = i * 256 + tid;          // oct index 0..511
            int row = p >> 2;
            int k8 = (p & 3) * 8;
            int gm = m0 + row; if (gm > M - 1) gm = M - 1;
            uint4 v = *(const uint4*)(A + offA + (long)gm * lda + (k0 + k8));
            *(uint4*)&At[row * LST + k8] = v;
        }
        // ---- stage B tile transposed in LDS: Bt[n][k] ----
        if (TRANSB) {
            // source [N,K] row-major -> b128 loads, b128 stores
            #pragma unroll
            for (int i = 0; i < 2; ++i) {
                int p = i * 256 + tid;
                int n = p >> 2;
                int k8 = (p & 3) * 8;
                int gn = n0 + n; if (gn > N - 1) gn = N - 1;
                uint4 v = *(const uint4*)(Bm + offB + (long)gn * ldb + (k0 + k8));
                *(uint4*)&Bt[n * LST + k8] = v;
            }
        } else {
            // source [K,N] row-major -> b128 loads, transpose via u16 stores
            #pragma unroll
            for (int i = 0; i < 2; ++i) {
                int p = i * 256 + tid;      // oct 0..511 (32 k-rows x 16 octs)
                int k = p >> 4;
                int n8 = (p & 15) * 8;
                int gnb = n0 + n8; if (gnb > N - 8) gnb = N - 8;
                U4x8 v;
                v.q = *(const uint4*)(Bm + offB + (long)(k0 + k) * ldb + gnb);
                #pragma unroll
                for (int j = 0; j < 8; ++j) Bt[(n8 + j) * LST + k] = v.s[j];
            }
        }
        // prefetch next K-chunk of A (speculative)
        if (k0 + TK < K) {
            int row = tid >> 1;
            int gm = m0 + row; if (gm > M - 1) gm = M - 1;
            __builtin_prefetch(A + offA + (long)gm * lda + (k0 + TK), 0, 1);
        }
        __syncthreads();

        // ---- fragments ----
        // A 16x32: lanes 0-15 row M=l, K={0..7,16..23}; lanes 16-31 row M=l-16, K={8..15,24..31}
        // B 32x16: lanes 0-15 col N=l, K={0..7,16..23}; lanes 16-31 col N=l-16, K={8..15,24..31}
        const int r = l & 15;
        const int half = (l >> 4) & 1;
        Frag16 a[4], b[2];
        #pragma unroll
        for (int mi = 0; mi < 4; ++mi) {
            const int abase = (mBase + mi * 16 + r) * LST;
            #pragma unroll
            for (int j = 0; j < 4; ++j) {
                int kk = half * 8 + 2 * j;
                a[mi].u[2 * j]         = At[abase + kk];
                a[mi].u[2 * j + 1]     = At[abase + kk + 1];
                a[mi].u[8 + 2 * j]     = At[abase + 16 + kk];
                a[mi].u[8 + 2 * j + 1] = At[abase + 16 + kk + 1];
            }
        }
        #pragma unroll
        for (int ni = 0; ni < 2; ++ni) {
            const int bbase = (nBase + ni * 16 + r) * LST + half * 8;
            #pragma unroll
            for (int i2 = 0; i2 < 16; ++i2) b[ni].u[i2] = Bt[bbase + (i2 & 7) + (i2 >> 3) * 16];
        }
        #pragma unroll
        for (int mi = 0; mi < 4; ++mi)
            #pragma unroll
            for (int ni = 0; ni < 2; ++ni)
                acc[mi][ni] = __builtin_amdgcn_wmma_f32_16x16x32_bf16(
                    false, a[mi].v, false, b[ni].v, (short)0, acc[mi][ni], false, false);
    }

    // ---- epilogue: lane l -> col = l&15, rows rh*8 + j ----
    const int col = l & 15;
    const int rh = (l >> 4) & 1;
    #pragma unroll
    for (int mi = 0; mi < 4; ++mi) {
        #pragma unroll
        for (int ni = 0; ni < 2; ++ni) {
            const int gn = n0 + nBase + ni * 16 + col;
            if (gn < N) {
                const float bv = bias ? bias[gn] : 0.f;
                #pragma unroll
                for (int j = 0; j < 8; ++j) {
                    int gm = m0 + mBase + mi * 16 + rh * 8 + j;
                    if (gm < M) {
                        float val = acc[mi][ni][j] * alpha + bv;
                        if (Cadd) val += Cadd[offAdd + (long)gm * ldadd + gn];
                        if (RELU) val = val > 0.f ? val : 0.f;
                        long ci = offC + (long)gm * ldc + gn;
                        if (OUTBF16) Cb[ci] = f2bf(val);
                        else         Cf[ci] = val;
                    }
                }
            }
        }
    }
}

// ---------------------------------------------------------------------------
// helpers / small kernels
// ---------------------------------------------------------------------------
__device__ __forceinline__ float block_reduce_sum(float v, float* sm) {
    int t = threadIdx.x;
    sm[t] = v; __syncthreads();
    for (int off = blockDim.x >> 1; off > 0; off >>= 1) {
        if (t < off) sm[t] += sm[t + off];
        __syncthreads();
    }
    float r = sm[0]; __syncthreads();
    return r;
}
__device__ __forceinline__ float block_reduce_max(float v, float* sm) {
    int t = threadIdx.x;
    sm[t] = v; __syncthreads();
    for (int off = blockDim.x >> 1; off > 0; off >>= 1) {
        if (t < off) sm[t] = fmaxf(sm[t], sm[t + off]);
        __syncthreads();
    }
    float r = sm[0]; __syncthreads();
    return r;
}

__global__ __launch_bounds__(256) void cvt_kernel(const float* __restrict__ src,
                                                  unsigned short* __restrict__ dst, long n) {
    long i = (long)blockIdx.x * blockDim.x + threadIdx.x;
    if (i < n) dst[i] = f2bf(src[i]);
}

__global__ __launch_bounds__(256) void embed_kernel(
    const int* __restrict__ raw_x, const int* __restrict__ deg,
    const float* __restrict__ vocab, const float* __restrict__ pe,
    const float* __restrict__ degemb, float* __restrict__ x)
{
    int tok = blockIdx.x;
    int s = tok % S_;
    int vid = raw_x[tok];
    int did = deg[tok];
    long base = (long)tok * D_;
    for (int d = threadIdx.x; d < D_; d += blockDim.x)
        x[base + d] = vocab[(long)vid * D_ + d] + pe[(long)s * D_ + d] + degemb[(long)did * D_ + d];
}

// pc_now/sib_now: b = arange(B), so blocks are independent; same-thread
// sequential adds make s_pc == s_sib collisions safe.
__global__ __launch_bounds__(256) void relnow_kernel(
    const int* __restrict__ pcn, const int* __restrict__ sibn,
    const float* __restrict__ arel, float* __restrict__ x)
{
    int blk = blockIdx.x;
    int b1 = pcn[blk * 2],  s1 = pcn[blk * 2 + 1];
    int b2 = sibn[blk * 2], s2 = sibn[blk * 2 + 1];
    for (int d = threadIdx.x; d < D_; d += blockDim.x) {
        x[((long)b1 * S_ + s1) * D_ + d] += arel[d];
        x[((long)b2 * S_ + s2) * D_ + d] += arel[D_ + d];
    }
}

// structural score bias, layer/head-invariant: bias[B,S,S] via atomics
__global__ __launch_bounds__(256) void relbias_kernel(
    const int* __restrict__ pc, const int* __restrict__ sib, int P,
    const float* __restrict__ relenc, float* __restrict__ biasS)
{
    int t = blockIdx.x * blockDim.x + threadIdx.x;
    if (t < P) {
        int b = pc[t * 3], s1 = pc[t * 3 + 1], s2 = pc[t * 3 + 2];
        atomicAdd(&biasS[((long)b * S_ + s1) * S_ + s2], relenc[0]);
        atomicAdd(&biasS[((long)b * S_ + s2) * S_ + s1], relenc[1]);
    } else if (t < 2 * P) {
        int i = t - P;
        int b = sib[i * 3], s1 = sib[i * 3 + 1], s2 = sib[i * 3 + 2];
        atomicAdd(&biasS[((long)b * S_ + s1) * S_ + s2], relenc[2]);
        atomicAdd(&biasS[((long)b * S_ + s2) * S_ + s1], relenc[2]);
    }
}

__global__ __launch_bounds__(256) void softmax_kernel(
    const float* __restrict__ scores, unsigned short* __restrict__ attnb)
{
    __shared__ float sm[256];
    long row = blockIdx.x;
    int t = threadIdx.x;
    float v = scores[row * S_ + t];
    float mx = block_reduce_max(v, sm);
    float p = __expf(v - mx);
    float s = block_reduce_sum(p, sm);
    attnb[row * S_ + t] = f2bf(p / s);
}

// x = LN(tmp + x); writes fp32 master and bf16 copy
__global__ __launch_bounds__(256) void ln_kernel(
    const float* __restrict__ tmp, float* __restrict__ x, unsigned short* __restrict__ xb,
    const float* __restrict__ g, const float* __restrict__ bt)
{
    __shared__ float sm[256];
    long base = (long)blockIdx.x * D_;
    int t = threadIdx.x;
    float v0 = tmp[base + t]       + x[base + t];
    float v1 = tmp[base + t + 256] + x[base + t + 256];
    float mean = block_reduce_sum(v0 + v1, sm) * (1.0f / D_);
    float d0 = v0 - mean, d1 = v1 - mean;
    float var = block_reduce_sum(d0 * d0 + d1 * d1, sm) * (1.0f / D_);
    float rstd = rsqrtf(var + 1e-5f);
    float y0 = d0 * rstd * g[t]       + bt[t];
    float y1 = d1 * rstd * g[t + 256] + bt[t + 256];
    x[base + t] = y0;        x[base + t + 256] = y1;
    xb[base + t] = f2bf(y0); xb[base + t + 256] = f2bf(y1);
}

// ---------------------------------------------------------------------------
// host side
// ---------------------------------------------------------------------------
static void gemm(hipStream_t st, bool transb, bool outbf, bool relu,
                 const unsigned short* A, const unsigned short* Bm,
                 const float* bias, const float* Cadd, float* Cf, unsigned short* Cb,
                 int M, int N, int K, int lda, int ldb, int ldc, int ldadd, int batch,
                 int divA, long sA0, long sA1,
                 int divB, long sB0, long sB1,
                 int divC, long sC0, long sC1,
                 int divAdd, long sAdd0, long sAdd1,
                 float alpha)
{
    dim3 g((M + TM - 1) / TM, (N + TN - 1) / TN, batch), b(256, 1, 1);
    if (transb)
        gemm_kernel<true, false, false><<<g, b, 0, st>>>(A, Bm, bias, Cadd, Cf, Cb, M, N, K,
            lda, ldb, ldc, ldadd, divA, sA0, sA1, divB, sB0, sB1, divC, sC0, sC1,
            divAdd, sAdd0, sAdd1, alpha);
    else if (outbf && relu)
        gemm_kernel<false, true, true><<<g, b, 0, st>>>(A, Bm, bias, Cadd, Cf, Cb, M, N, K,
            lda, ldb, ldc, ldadd, divA, sA0, sA1, divB, sB0, sB1, divC, sC0, sC1,
            divAdd, sAdd0, sAdd1, alpha);
    else if (outbf)
        gemm_kernel<false, true, false><<<g, b, 0, st>>>(A, Bm, bias, Cadd, Cf, Cb, M, N, K,
            lda, ldb, ldc, ldadd, divA, sA0, sA1, divB, sB0, sB1, divC, sC0, sC1,
            divAdd, sAdd0, sAdd1, alpha);
    else
        gemm_kernel<false, false, false><<<g, b, 0, st>>>(A, Bm, bias, Cadd, Cf, Cb, M, N, K,
            lda, ldb, ldc, ldadd, divA, sA0, sA1, divB, sB0, sB1, divC, sC0, sC1,
            divAdd, sAdd0, sAdd1, alpha);
}

static void cvt(hipStream_t st, const float* src, unsigned short* dst, long n) {
    cvt_kernel<<<(unsigned)((n + 255) / 256), 256, 0, st>>>(src, dst, n);
}

extern "C" void kernel_launch(void* const* d_in, const int* in_sizes, int n_in,
                              void* d_out, int out_size, void* d_ws, size_t ws_size,
                              hipStream_t stream) {
    (void)n_in; (void)out_size; (void)ws_size;
    const int*   raw_x  = (const int*)d_in[0];
    const int*   deg    = (const int*)d_in[1];
    const int*   pc     = (const int*)d_in[2];
    const int*   pcn    = (const int*)d_in[3];
    const int*   sib    = (const int*)d_in[4];
    const int*   sibn   = (const int*)d_in[5];
    const float* vocab  = (const float*)d_in[6];
    const float* pe     = (const float*)d_in[7];
    const float* degemb = (const float*)d_in[8];
    const float* relenc = (const float*)d_in[9];
    const float* arel   = (const float*)d_in[10];
    const float* Wq = (const float*)d_in[11]; const float* bq = (const float*)d_in[12];
    const float* Wk = (const float*)d_in[13]; const float* bk = (const float*)d_in[14];
    const float* Wv = (const float*)d_in[15]; const float* bv = (const float*)d_in[16];
    const float* Wo = (const float*)d_in[17]; const float* bo = (const float*)d_in[18];
    const float* W1 = (const float*)d_in[19]; const float* b1 = (const float*)d_in[20];
    const float* W2 = (const float*)d_in[21]; const float* b2 = (const float*)d_in[22];
    const float* lng = (const float*)d_in[23]; const float* lnb = (const float*)d_in[24];
    const float* Wlog = (const float*)d_in[25];
    const int P = in_sizes[2] / 3;

    // ---- workspace bump allocator ----
    char* wp = (char*)d_ws;
    auto alloc = [&](size_t bytes) -> void* {
        void* p = (void*)wp;
        wp += (bytes + 255) & ~(size_t)255;
        return p;
    };
    float*          x      = (float*)alloc((size_t)NTOK * D_ * 4);
    float*          tmp    = (float*)alloc((size_t)NTOK * D_ * 4);
    unsigned short* xb     = (unsigned short*)alloc((size_t)NTOK * D_ * 2);
    unsigned short* qb     = (unsigned short*)alloc((size_t)NTOK * D_ * 2);
    unsigned short* kb     = (unsigned short*)alloc((size_t)NTOK * D_ * 2);
    unsigned short* vb     = (unsigned short*)alloc((size_t)NTOK * D_ * 2);
    unsigned short* ob     = (unsigned short*)alloc((size_t)NTOK * D_ * 2);
    float*          scoresW= (float*)alloc((size_t)B_ * H_ * S_ * S_ * 4);
    unsigned short* bigb   = (unsigned short*)alloc((size_t)B_ * H_ * S_ * S_ * 2); // attn probs / ffn hidden
    float*          biasS  = (float*)alloc((size_t)B_ * S_ * S_ * 4);
    unsigned short* Wqb    = (unsigned short*)alloc((size_t)L_ * D_ * D_ * 2);
    unsigned short* Wkb    = (unsigned short*)alloc((size_t)L_ * D_ * D_ * 2);
    unsigned short* Wvb    = (unsigned short*)alloc((size_t)L_ * D_ * D_ * 2);
    unsigned short* Wob    = (unsigned short*)alloc((size_t)L_ * D_ * D_ * 2);
    unsigned short* W1b    = (unsigned short*)alloc((size_t)L_ * D_ * DFF_ * 2);
    unsigned short* W2b    = (unsigned short*)alloc((size_t)L_ * DFF_ * D_ * 2);
    unsigned short* Wlogb  = (unsigned short*)alloc((size_t)D_ * V_ * 2);

    // ---- one-time per launch: structural bias + bf16 weights ----
    hipMemsetAsync(biasS, 0, (size_t)B_ * S_ * S_ * 4, stream);
    relbias_kernel<<<(2 * P + 255) / 256, 256, 0, stream>>>(pc, sib, P, relenc, biasS);
    cvt(stream, Wq, Wqb, (long)L_ * D_ * D_);
    cvt(stream, Wk, Wkb, (long)L_ * D_ * D_);
    cvt(stream, Wv, Wvb, (long)L_ * D_ * D_);
    cvt(stream, Wo, Wob, (long)L_ * D_ * D_);
    cvt(stream, W1, W1b, (long)L_ * D_ * DFF_);
    cvt(stream, W2, W2b, (long)L_ * DFF_ * D_);
    cvt(stream, Wlog, Wlogb, (long)D_ * V_);

    // ---- embedding ----
    embed_kernel<<<NTOK, 256, 0, stream>>>(raw_x, deg, vocab, pe, degemb, x);
    relnow_kernel<<<B_, 256, 0, stream>>>(pcn, sibn, arel, x);
    cvt(stream, x, xb, (long)NTOK * D_);

    const long SD = (long)S_ * D_;
    const long SS = (long)S_ * S_;

    for (int i = 0; i < L_; ++i) {
        // Q,K,V projections -> bf16
        gemm(stream, false, true, false, xb, Wqb + (size_t)i * D_ * D_, bq + (size_t)i * D_,
             nullptr, nullptr, qb, NTOK, D_, D_, D_, D_, D_, 1, 1,
             1, 0, 0, 1, 0, 0, 1, 0, 0, 1, 0, 0, 1.0f);
        gemm(stream, false, true, false, xb, Wkb + (size_t)i * D_ * D_, bk + (size_t)i * D_,
             nullptr, nullptr, kb, NTOK, D_, D_, D_, D_, D_, 1, 1,
             1, 0, 0, 1, 0, 0, 1, 0, 0, 1, 0, 0, 1.0f);
        gemm(stream, false, true, false, xb, Wvb + (size_t)i * D_ * D_, bv + (size_t)i * D_,
             nullptr, nullptr, vb, NTOK, D_, D_, D_, D_, D_, 1, 1,
             1, 0, 0, 1, 0, 0, 1, 0, 0, 1, 0, 0, 1.0f);
        // scores = q @ k^T * 1/sqrt(64) + structural bias  (batched over B*H)
        gemm(stream, true, false, false, qb, kb, nullptr, biasS, scoresW, nullptr,
             S_, S_, DH_, D_, D_, S_, S_, B_ * H_,
             H_, SD, 64, H_, SD, 64, 1, SS, 0, H_, SS, 0, 0.125f);
        // softmax -> bf16 probs
        softmax_kernel<<<B_ * H_ * S_, 256, 0, stream>>>(scoresW, bigb);
        // o = attn @ v -> bf16 in [B,S,D] (head-interleaved) layout
        gemm(stream, false, true, false, bigb, vb, nullptr, nullptr, nullptr, ob,
             S_, DH_, S_, S_, D_, D_, 1, B_ * H_,
             1, SS, 0, H_, SD, 64, H_, SD, 64, 1, 0, 0, 1.0f);
        // O projection -> fp32 tmp
        gemm(stream, false, false, false, ob, Wob + (size_t)i * D_ * D_, bo + (size_t)i * D_,
             nullptr, tmp, nullptr, NTOK, D_, D_, D_, D_, D_, 1, 1,
             1, 0, 0, 1, 0, 0, 1, 0, 0, 1, 0, 0, 1.0f);
        // x = LN(tmp + x)
        ln_kernel<<<NTOK, 256, 0, stream>>>(tmp, x, xb, lng + (size_t)i * D_, lnb + (size_t)i * D_);
        // FFN1: relu(x @ W1 + b1) -> bf16
        gemm(stream, false, true, true, xb, W1b + (size_t)i * D_ * DFF_, b1 + (size_t)i * DFF_,
             nullptr, nullptr, bigb, NTOK, DFF_, D_, D_, DFF_, DFF_, 1, 1,
             1, 0, 0, 1, 0, 0, 1, 0, 0, 1, 0, 0, 1.0f);
        // FFN2 -> fp32 tmp
        gemm(stream, false, false, false, bigb, W2b + (size_t)i * DFF_ * D_, b2 + (size_t)i * D_,
             nullptr, tmp, nullptr, NTOK, D_, DFF_, DFF_, D_, D_, 1, 1,
             1, 0, 0, 1, 0, 0, 1, 0, 0, 1, 0, 0, 1.0f);
        // x = LN(tmp + x)
        ln_kernel<<<NTOK, 256, 0, stream>>>(tmp, x, xb, lng + (size_t)i * D_, lnb + (size_t)i * D_);
    }

    // logits = x[:, -1, :] @ W_logits  -> d_out fp32 [B, V]
    gemm(stream, false, false, false, xb + (size_t)(S_ - 1) * D_, Wlogb, nullptr, nullptr,
         (float*)d_out, nullptr, B_, V_, D_, S_ * D_, V_, V_, 1, 1,
         1, 0, 0, 1, 0, 0, 1, 0, 0, 1, 0, 0, 1.0f);
}
